// MySelfAttention2_71880572666095
// MI455X (gfx1250) — matrix operand order repaired and checked by reading, not compiled, not run on hardware
//
#include <hip/hip_runtime.h>

// B=8, S=2048, D=512, O=512. fp32 in/out; f16 WMMA internally.
// Workspace (needs 3*16384*512*2 + 3*512*512*2 = 51.9 MB):
//   q_h : _Float16[16384*512]      row-major (pre-scaled by 1/sqrt(512))  at ws + 0
//   k_h : _Float16[16384*512]      row-major                              at ws + 8388608
//   v_t : _Float16[8][512][2048]   TRANSPOSED per batch                   at ws + 16777216
//   w_t : _Float16[3][512][512]    W transposed ([m][o][d]) f16           at ws + 25165824

typedef __attribute__((ext_vector_type(16))) _Float16 v16h;
typedef __attribute__((ext_vector_type(8)))  float    v8f;
typedef __attribute__((ext_vector_type(2)))  __fp16   fp16x2;
typedef __attribute__((ext_vector_type(4)))  unsigned ui4;
typedef __attribute__((ext_vector_type(8)))  unsigned ui8;

__device__ __forceinline__ v8f wmma_f16(v16h a, v16h b, v8f c) {
  return __builtin_amdgcn_wmma_f32_16x16x32_f16(
      false, a, false, b, (short)0, c, false, false);
}

// A-matrix 16x32 f16 (ISA 7.12.2): lane=(h<<4)|row; VGPR j: j<4 -> K=2j(+8h), j>=4 -> K=16+2(j-4)(+8h)
__device__ __forceinline__ v16h load_fragA(const _Float16* base, int rc, int stride,
                                           int k0, int h) {
  union { v16h v; unsigned u[8]; } f;
  const _Float16* p = base + rc * stride + k0;
#pragma unroll
  for (int j = 0; j < 8; ++j) {
    int kp = ((j < 4) ? (2 * j) : (16 + 2 * (j - 4))) + 8 * h;
    f.u[j] = *(const unsigned*)(p + kp);
  }
  return f.v;
}

// B-matrix 32x16 f16: lane=(h<<4)|col; VGPR j: j<4 -> K=2j(+8h), j>=4 -> K=16+2(j-4)(+8h). Source stored [col][k].
__device__ __forceinline__ v16h load_fragB(const _Float16* base, int rc, int stride,
                                           int k0, int h) {
  union { v16h v; unsigned u[8]; } f;
  const _Float16* p = base + rc * stride + k0;
#pragma unroll
  for (int j = 0; j < 8; ++j) {
    int kp = ((j < 4) ? (2 * j) : (16 + 2 * (j - 4))) + 8 * h;
    f.u[j] = *(const unsigned*)(p + kp);
  }
  return f.v;
}

// ---- CDNA5 async global->LDS copy (ASYNCcnt), 16B per lane ----
__device__ __forceinline__ unsigned lds_off_u32(const void* p) {
  return (unsigned)(unsigned long long)(uintptr_t)p;  // low 32 bits = LDS byte offset
}
__device__ __forceinline__ void async_b128(unsigned lds_byte_off, const void* gptr) {
  unsigned long long ga = (unsigned long long)(uintptr_t)gptr;
  asm volatile("global_load_async_to_lds_b128 %0, %1, off"
               :: "v"(lds_byte_off), "v"(ga) : "memory");
}
__device__ __forceinline__ void wait_async0() {
  asm volatile("s_wait_asynccnt 0x0" ::: "memory");
}

// ---- CDNA5 Tensor Data Mover: 2D tile global->LDS with LDS pad insertion ----
// D# per ISA 08_async_tensor §8.3/8.4. Issued once per (uniform) call; TENSORcnt.
__device__ __forceinline__ void tdm_stage(unsigned lds_byte, const void* gptr,
                                          unsigned tile_d0, unsigned tile_d1,
                                          unsigned tensor_d0, unsigned tensor_d1,
                                          unsigned stride_d0,
                                          unsigned pad_interval, unsigned pad_amount) {
  unsigned long long ga = (unsigned long long)(uintptr_t)gptr;
  ui4 g0;
  g0.x = 1u;                                   // count=1, user descriptor
  g0.y = lds_byte;                             // lds_addr
  g0.z = (unsigned)ga;                         // global_addr[31:0]
  g0.w = (unsigned)(ga >> 32) | (2u << 30);    // global_addr[56:32] | type=2 ("image")
  ui8 g1;
  g1.s0 = (1u << 16)                           // data_size = 1 (2 bytes)
        | (1u << 20)                           // pad_enable
        | (pad_interval << 22) | (pad_amount << 25);
  g1.s1 = (tensor_d0 & 0xFFFFu) << 16;         // tensor_dim0[15:0]
  g1.s2 = (tensor_d0 >> 16) | ((tensor_d1 & 0xFFFFu) << 16);
  g1.s3 = (tensor_d1 >> 16) | (tile_d0 << 16); // tile_dim0
  g1.s4 = tile_d1;                             // tile_dim1 (tile_dim2 = 0)
  g1.s5 = stride_d0;                           // tensor_dim0_stride[31:0]
  g1.s6 = 0u;
  g1.s7 = 0u;
  asm volatile("tensor_load_to_lds %0, %1" :: "s"(g0), "s"(g1) : "memory");
}
__device__ __forceinline__ void wait_tensor0() {
  __builtin_amdgcn_s_wait_tensorcnt(0);
}

// ---------------------------------------------------------------------------
// Kernel 0: one-time W fp32 -> f16 transposed ([m][o][d]).
// ---------------------------------------------------------------------------
__global__ __launch_bounds__(256) void wconv_kernel(const float* __restrict__ W,
                                                    _Float16* __restrict__ wt) {
  const int m = blockIdx.y;
  const float* Wm = W + (size_t)m * 262144;
  _Float16* wtm = wt + (size_t)m * 262144;
  int i = (blockIdx.x * 256 + threadIdx.x) * 4;
  int d = i >> 9, o = i & 511;
  float4 v = *(const float4*)(Wm + (size_t)d * 512 + o);
  wtm[(size_t)(o + 0) * 512 + d] = (_Float16)v.x;
  wtm[(size_t)(o + 1) * 512 + d] = (_Float16)v.y;
  wtm[(size_t)(o + 2) * 512 + d] = (_Float16)v.z;
  wtm[(size_t)(o + 3) * 512 + d] = (_Float16)v.w;
}

// ---------------------------------------------------------------------------
// Kernel 1: projections. grid=(8, 256, 3), block=256 (8 waves).
// X staged fp32->f16 with packed cvt; W^T tiles staged via async b128.
// m==0 output pre-scaled by 1/sqrt(512); m==2 written transposed per batch.
// ---------------------------------------------------------------------------
__global__ __launch_bounds__(256) void proj_kernel(const float* __restrict__ qv,
                                                   const float* __restrict__ kv,
                                                   const float* __restrict__ vv,
                                                   const _Float16* __restrict__ wt,
                                                   _Float16* __restrict__ ws) {
  const int m = blockIdx.z;
  const float* X = (m == 0) ? qv : ((m == 1) ? kv : vv);
  const _Float16* wtm = wt + (size_t)m * 262144;
  _Float16* out = ws + (size_t)m * 16384 * 512;

  const int Mb = blockIdx.y * 64;
  const int Nb = blockIdx.x * 64;
  const int tid = threadIdx.x;
  const int w = tid >> 5, ln = tid & 15, h = (tid >> 4) & 1;
  const int rs = w >> 1;
  const int ct0 = (w & 1) * 2;

  __shared__ _Float16 smem[5120];   // xs = [0..2559] (64x40), wts = [2560..] (64x40)
  _Float16* xs  = smem;
  _Float16* wts = smem + 2560;
  const unsigned wts_lds = lds_off_u32(wts);

  const v8f vzero = {0.f, 0.f, 0.f, 0.f, 0.f, 0.f, 0.f, 0.f};
  v8f acc0 = vzero, acc1 = vzero;

  for (int c = 0; c < 16; ++c) {
    const int d0 = c * 32;
    // async-stage W^T tile: 64 cols x 32 d -> wts (stride 40 halves / 80 B)
    {
      int col = tid >> 2, ch = tid & 3;
      async_b128(wts_lds + col * 80 + ch * 16,
                 wtm + (size_t)(Nb + col) * 512 + d0 + ch * 8);
    }
    // stage X 64x32 fp32 -> f16 (float4 loads, packed cvt, b64 LDS stores)
#pragma unroll
    for (int n = 0; n < 2; ++n) {
      int i = tid + n * 256;
      int rl = i >> 3, dq = (i & 7) * 4;
      float4 xv = *(const float4*)(X + (size_t)(Mb + rl) * 512 + d0 + dq);
      union { fp16x2 hh[2]; uint2 u; } cv;
      cv.hh[0] = __builtin_amdgcn_cvt_pkrtz(xv.x, xv.y);
      cv.hh[1] = __builtin_amdgcn_cvt_pkrtz(xv.z, xv.w);
      *(uint2*)(xs + rl * 40 + dq) = cv.u;
    }
    wait_async0();
    __syncthreads();

    v16h a  = load_fragA(xs, rs * 16 + ln, 40, 0, h);
    v16h b0 = load_fragB(wts, (ct0 + 0) * 16 + ln, 40, 0, h);
    v16h b1 = load_fragB(wts, (ct0 + 1) * 16 + ln, 40, 0, h);
    acc0 = wmma_f16(a, b0, acc0);
    acc1 = wmma_f16(a, b1, acc1);
    __syncthreads();
  }

  if (m == 2) {
    // V: transposed per batch: out[b][col][s]; 8 contiguous halves -> one b128/lane
    const int bb = Mb >> 11;
    const int s0 = (Mb & 2047) + rs * 16 + 8 * h;
#pragma unroll
    for (int t = 0; t < 2; ++t) {
      v8f acc = t ? acc1 : acc0;
      union { _Float16 hh[8]; uint4 q; } pk;
#pragma unroll
      for (int r = 0; r < 8; ++r) pk.hh[r] = (_Float16)acc[r];
      int col = Nb + (ct0 + t) * 16 + ln;
      *(uint4*)(out + (size_t)bb * 1048576 + (size_t)col * 2048 + s0) = pk.q;
    }
  } else {
    // q/k: bounce through LDS for coalesced b32 row stores; q pre-scaled
    const float sc = (m == 0) ? 0.04419417382f : 1.0f;
#pragma unroll
    for (int t = 0; t < 2; ++t) {
      v8f acc = t ? acc1 : acc0;
      int col = (ct0 + t) * 16 + ln;
#pragma unroll
      for (int r = 0; r < 8; ++r)
        smem[(rs * 16 + r + 8 * h) * 72 + col] = (_Float16)(acc[r] * sc);
    }
    __syncthreads();
#pragma unroll
    for (int n = 0; n < 8; ++n) {
      int i = tid + n * 256;
      int row = i >> 5, cu = i & 31;
      *(unsigned*)(out + (size_t)(Mb + row) * 512 + Nb + cu * 2) =
          *(const unsigned*)(smem + row * 72 + cu * 2);
    }
  }
}

// ---------------------------------------------------------------------------
// Kernel 2: flash attention. grid=(64, 8), block=256 (8 waves).
// K and V^T tiles staged by the Tensor Data Mover (with LDS pad insertion);
// V^T DMA overlaps the softmax. Scores combined with a two-phase store/add.
// ---------------------------------------------------------------------------
__global__ __launch_bounds__(256) void attn_kernel(const _Float16* __restrict__ qh,
                                                   const _Float16* __restrict__ kh,
                                                   const _Float16* __restrict__ vt,
                                                   float* __restrict__ out) {
  const int b = blockIdx.y;
  const int q0 = blockIdx.x * 32;
  const int bbase = b * 2048;
  const int tid = threadIdx.x;
  const int w = tid >> 5, ln = tid & 15, h = (tid >> 4) & 1;

  __shared__ _Float16 kvbuf[20480];   // K: [32][520] halves | V^T: [512][40] halves
  __shared__ float    sbuf[32 * 33];
  __shared__ _Float16 pbuf[32 * 40];
  __shared__ float    mrow[32], lrow[32], arow[32];

  const int tile = w & 3, rs = tile >> 1, ks = tile & 1, dh = w >> 2;
  const unsigned kv_lds = lds_off_u32(kvbuf);

  // preload this wave's Q fragments (16 rows x 256-wide D-half); q pre-scaled
  v16h qf[8];
#pragma unroll
  for (int c = 0; c < 8; ++c)
    qf[c] = load_fragA(qh, bbase + q0 + rs * 16 + ln, 512, dh * 256 + c * 32, h);

  const v8f vzero = {0.f, 0.f, 0.f, 0.f, 0.f, 0.f, 0.f, 0.f};
  v8f accO[8];
#pragma unroll
  for (int i = 0; i < 8; ++i) accO[i] = vzero;

  if (tid < 32) { mrow[tid] = -1e30f; lrow[tid] = 0.f; }
  __syncthreads();

  for (int kt = 0; kt < 64; ++kt) {
    const int key0 = kt * 32;

    // ---- TDM: K tile [32 keys][512 d], row 1024B + 16B pad -> stride 520 halves ----
    if (w == 0) {
      tdm_stage(kv_lds, kh + (size_t)(bbase + key0) * 512,
                /*tile_d0=*/512, /*tile_d1=*/32,
                /*tensor_d0=*/512, /*tensor_d1=*/2048, /*stride_d0=*/512,
                /*pad_interval: 256 dw=*/7, /*pad_amount: 4 dw=*/3);
      wait_tensor0();
    }
    __syncthreads();

    // ---- partial scores over this wave's D-half ----
    v8f sacc = vzero;
#pragma unroll
    for (int c = 0; c < 8; ++c) {
      v16h bf = load_fragB(kvbuf, ks * 16 + ln, 520, dh * 256 + c * 32, h);
      sacc = wmma_f16(qf[c], bf, sacc);
    }
    // two-phase combine of the two D-halves (no atomics, no zeroing)
    if (dh == 0) {
#pragma unroll
      for (int r = 0; r < 8; ++r)
        sbuf[(rs * 16 + r + 8 * h) * 33 + ks * 16 + ln] = sacc[r];
    }
    __syncthreads();
    if (dh == 1) {
#pragma unroll
      for (int r = 0; r < 8; ++r)
        sbuf[(rs * 16 + r + 8 * h) * 33 + ks * 16 + ln] += sacc[r];
    }
    __syncthreads();

    // ---- TDM: V^T tile [512 cols][32 keys], row 64B + 16B pad -> stride 40 halves;
    //      DMA overlaps the softmax (both on wave 0) ----
    if (w == 0) {
      tdm_stage(kv_lds, vt + (size_t)b * 1048576 + key0,
                /*tile_d0=*/32, /*tile_d1=*/512,
                /*tensor_d0=*/2048, /*tensor_d1=*/512, /*stride_d0=*/2048,
                /*pad_interval: 16 dw=*/3, /*pad_amount: 4 dw=*/3);
      // online softmax (threads 0..31) while the DMA runs
      float mold = mrow[tid];
      float mt = -1e30f;
#pragma unroll
      for (int k = 0; k < 32; ++k) mt = fmaxf(mt, sbuf[tid * 33 + k]);
      float mn = fmaxf(mold, mt);
      float corr = __expf(mold - mn);
      float sum = 0.f;
#pragma unroll
      for (int k = 0; k < 32; ++k) {
        float p = __expf(sbuf[tid * 33 + k] - mn);
        pbuf[tid * 40 + k] = (_Float16)p;
        sum += p;
      }
      mrow[tid] = mn;
      arow[tid] = corr;
      lrow[tid] = corr * lrow[tid] + sum;
      wait_tensor0();
    }
    __syncthreads();

    // ---- rescale O, accumulate P*V for this wave's 64-col slice ----
#pragma unroll
    for (int rt = 0; rt < 2; ++rt) {
      float al[8];
#pragma unroll
      for (int r = 0; r < 8; ++r) al[r] = arow[rt * 16 + r + 8 * h];
      v16h pa = load_fragA(pbuf, rt * 16 + ln, 40, 0, h);
#pragma unroll
      for (int ct = 0; ct < 4; ++ct) {
        v8f acc = accO[rt * 4 + ct];
#pragma unroll
        for (int r = 0; r < 8; ++r) acc[r] *= al[r];
        v16h vb = load_fragB(kvbuf, w * 64 + ct * 16 + ln, 40, 0, h);
        accO[rt * 4 + ct] = wmma_f16(pa, vb, acc);
      }
    }
    if (kt < 63)
      __builtin_prefetch(kh + (size_t)(bbase + key0 + 32 + (tid & 31)) * 512 +
                             (tid >> 5) * 64, 0, 0);
    __syncthreads();
  }

  // ---- epilogue: O / l ----
#pragma unroll
  for (int rt = 0; rt < 2; ++rt) {
    float il[8];
#pragma unroll
    for (int r = 0; r < 8; ++r) il[r] = 1.f / lrow[rt * 16 + r + 8 * h];
#pragma unroll
    for (int ct = 0; ct < 4; ++ct) {
      int col = w * 64 + ct * 16 + ln;
#pragma unroll
      for (int r = 0; r < 8; ++r) {
        int row = bbase + q0 + rt * 16 + r + 8 * h;
        out[(size_t)row * 512 + col] = accO[rt * 4 + ct][r] * il[r];
      }
    }
  }
}

extern "C" void kernel_launch(void* const* d_in, const int* in_sizes, int n_in,
                              void* d_out, int out_size, void* d_ws, size_t ws_size,
                              hipStream_t stream) {
  (void)in_sizes; (void)n_in; (void)out_size; (void)ws_size;
  const float* q_v = (const float*)d_in[0];
  const float* k_v = (const float*)d_in[1];
  const float* v_v = (const float*)d_in[2];
  const float* W   = (const float*)d_in[3];

  _Float16* ws = (_Float16*)d_ws;          // needs >= 52 MB
  _Float16* qh = ws;
  _Float16* kh = ws + (size_t)8388608;
  _Float16* vt = ws + (size_t)16777216;
  _Float16* wt = ws + (size_t)25165824;

  wconv_kernel<<<dim3(256, 3), 256, 0, stream>>>(W, wt);
  proj_kernel<<<dim3(8, 256, 3), 256, 0, stream>>>(q_v, k_v, v_v, wt, ws);
  attn_kernel<<<dim3(64, 8), 256, 0, stream>>>(qh, kh, vt, (float*)d_out);
}
